// GAT_12618613915871
// MI455X (gfx1250) — matrix operand order repaired and checked by reading, not compiled, not run on hardware
//
#include <hip/hip_runtime.h>
#include <hip/hip_fp16.h>

#define DD 64
#define NEG_SLOPE 0.2f
#define BN_EPS 1e-5f
#define SM_EPS 1e-9f

typedef __attribute__((ext_vector_type(16))) _Float16 v16h;
typedef __attribute__((ext_vector_type(8)))  _Float16 v8h;
typedef __attribute__((ext_vector_type(8)))  float    v8f;

// order-preserving float <-> uint mapping for atomicMax on floats
__device__ __forceinline__ unsigned fmax_map(float f) {
  unsigned u = __float_as_uint(f);
  return (u & 0x80000000u) ? ~u : (u | 0x80000000u);
}
__device__ __forceinline__ float fmax_unmap(unsigned u) {
  return (u & 0x80000000u) ? __uint_as_float(u ^ 0x80000000u) : __uint_as_float(~u);
}

// ---------- prep: fp32 -> fp16 elementwise ----------
__global__ void k_f32_to_f16(const float* __restrict__ x, _Float16* __restrict__ y, int n) {
  int i = blockIdx.x * blockDim.x + threadIdx.x;
  if (i < n) y[i] = (_Float16)x[i];
}

// ---------- prep: per-layer weight fp16 conversion + v_e = fc_e_w^T @ attn_e ----------
__global__ void k_prep_weights(const float* __restrict__ w0, const float* __restrict__ we0, const float* __restrict__ ae0,
                               const float* __restrict__ w1, const float* __restrict__ we1, const float* __restrict__ ae1,
                               const float* __restrict__ w2, const float* __restrict__ we2, const float* __restrict__ ae2,
                               _Float16* __restrict__ W16, float* __restrict__ ve) {
  int l = blockIdx.x;
  const float* W  = (l == 0) ? w0  : (l == 1) ? w1  : w2;
  const float* We = (l == 0) ? we0 : (l == 1) ? we1 : we2;
  const float* Ae = (l == 0) ? ae0 : (l == 1) ? ae1 : ae2;
  for (int i = threadIdx.x; i < DD * DD; i += blockDim.x)
    W16[l * DD * DD + i] = (_Float16)W[i];
  if (threadIdx.x < DD) {
    int k = threadIdx.x;
    float acc = 0.f;
    for (int j = 0; j < DD; ++j) acc += We[j * DD + k] * Ae[j];
    ve[l * DD + k] = acc;
  }
}

// ---------- ee for all 3 layers in one pass over edge_x: one wave per edge ----------
__global__ void k_ee(const float* __restrict__ edge_x, const float* __restrict__ ve,
                     float* __restrict__ ee, int E_) {
  int wave = (blockIdx.x * blockDim.x + threadIdx.x) >> 5;
  if (wave >= E_) return;
  int lane = threadIdx.x & 31;
  float2 x = *(const float2*)(edge_x + (size_t)wave * DD + lane * 2);
  float a0 = x.x * ve[lane * 2] + x.y * ve[lane * 2 + 1];
  float a1 = x.x * ve[DD + lane * 2] + x.y * ve[DD + lane * 2 + 1];
  float a2 = x.x * ve[2 * DD + lane * 2] + x.y * ve[2 * DD + lane * 2 + 1];
  for (int o = 16; o > 0; o >>= 1) {
    a0 += __shfl_xor(a0, o, 32);
    a1 += __shfl_xor(a1, o, 32);
    a2 += __shfl_xor(a2, o, 32);
  }
  if (lane == 0) {
    ee[wave] = a0;
    ee[E_ + wave] = a1;
    ee[2 * E_ + wave] = a2;
  }
}

// ---------- WMMA GEMM: hs[16x64 tile] = h16[16x64] @ W16^T, one wave per row tile ----------
__global__ void k_gemm_wmma(const _Float16* __restrict__ X, const _Float16* __restrict__ W16,
                            float* __restrict__ hs, int nRows, int nTiles) {
  int wave = (blockIdx.x * blockDim.x + threadIdx.x) >> 5;
  if (wave >= nTiles) return;          // wave-uniform branch: EXEC stays all-ones inside
  int lane = threadIdx.x & 31;
  int half = lane >> 4;                // 0 or 1
  int l15  = lane & 15;
  int rowBase = wave * 16;

  v8f acc[4] = {v8f{}, v8f{}, v8f{}, v8f{}};

#pragma unroll
  for (int s = 0; s < 2; ++s) {        // K steps: K = 32*s .. 32*s+31
    // A fragment: 16x32 f16. Lane half h holds K chunks [h*8, h*8+8) and [h*8+16, h*8+24)
    int arow = rowBase + l15;
    int ac = (half << 3) + (s << 5);
    union { v16h v; v8h h[2]; } A;
    A.h[0] = v8h{};
    A.h[1] = v8h{};
    if (arow < nRows) {
      A.h[0] = *(const v8h*)(X + (size_t)arow * DD + ac);
      A.h[1] = *(const v8h*)(X + (size_t)arow * DD + ac + 16);
    }
#pragma unroll
    for (int t = 0; t < 4; ++t) {      // 4 column tiles of 16
      // B fragment: 32x16 f16, B[k][n] = W[n][k]; lane half h holds K chunks [h*8, h*8+8) and [h*8+16, h*8+24) of column n
      int n  = l15 + (t << 4);
      int bk = (half << 3) + (s << 5);
      union { v16h v; v8h h[2]; } B;
      B.h[0] = *(const v8h*)(W16 + (size_t)n * DD + bk);
      B.h[1] = *(const v8h*)(W16 + (size_t)n * DD + bk + 16);
      acc[t] = __builtin_amdgcn_wmma_f32_16x16x32_f16(
          false, A.v, false, B.v, (short)0, acc[t], false, false);
    }
  }

  // C layout: VGPR r -> rows M=r (lanes 0-15) / M=r+8 (lanes 16-31), col N = l15 + 16t
#pragma unroll
  for (int t = 0; t < 4; ++t) {
#pragma unroll
    for (int r = 0; r < 8; ++r) {
      int row = rowBase + (half << 3) + r;
      if (row < nRows) hs[(size_t)row * DD + (t << 4) + l15] = acc[t][r];
    }
  }
}

// ---------- el/er: one wave per node, dot(hs[n], attn_l/attn_r) ----------
__global__ void k_eler(const float* __restrict__ hs, const float* __restrict__ al,
                       const float* __restrict__ ar, float* __restrict__ el,
                       float* __restrict__ er, int N_) {
  int wave = (blockIdx.x * blockDim.x + threadIdx.x) >> 5;
  if (wave >= N_) return;
  int lane = threadIdx.x & 31;
  float2 h2 = *(const float2*)(hs + (size_t)wave * DD + lane * 2);
  float2 a2 = *(const float2*)(al + lane * 2);
  float2 b2 = *(const float2*)(ar + lane * 2);
  float sl = h2.x * a2.x + h2.y * a2.y;
  float sr = h2.x * b2.x + h2.y * b2.y;
  for (int o = 16; o > 0; o >>= 1) {
    sl += __shfl_xor(sl, o, 32);
    sr += __shfl_xor(sr, o, 32);
  }
  if (lane == 0) { el[wave] = sl; er[wave] = sr; }
}

// ---------- per-layer init ----------
__global__ void k_init(float* __restrict__ out, unsigned* __restrict__ m,
                       float* __restrict__ denom, float* __restrict__ colsum,
                       float* __restrict__ colsq, int N_, int total) {
  int i = blockIdx.x * blockDim.x + threadIdx.x;
  if (i < total) out[i] = 0.f;
  if (i < N_) { m[i] = 0u; denom[i] = 0.f; }
  if (i < DD) { colsum[i] = 0.f; colsq[i] = 0.f; }
}

// ---------- logits + segment max (atomic on order-preserving uints) ----------
__global__ void k_logits(const float* __restrict__ el, const float* __restrict__ er,
                         const float* __restrict__ ee, const int* __restrict__ src,
                         const int* __restrict__ dst, float* __restrict__ logits,
                         unsigned* __restrict__ m, int E_) {
  int e = blockIdx.x * blockDim.x + threadIdx.x;
  if (e >= E_) return;
  float lg = el[src[e]] + er[dst[e]] + ee[e];
  lg = (lg > 0.f) ? lg : NEG_SLOPE * lg;
  logits[e] = lg;
  atomicMax(&m[dst[e]], fmax_map(lg));
}

// ---------- aggregate: out[dst] += exp(lg - m)*hs[src], denom[dst] += exp ----------
__global__ void k_aggregate(const float* __restrict__ hs, const float* __restrict__ logits,
                            const unsigned* __restrict__ m, const int* __restrict__ src,
                            const int* __restrict__ dst, float* __restrict__ out,
                            float* __restrict__ denom, int E_) {
  int t = blockIdx.x * blockDim.x + threadIdx.x;
  int e = t >> 6;
  int j = t & 63;
  if (e >= E_) return;
  int s = src[e], d = dst[e];
  if (j == 0) {
    int e2 = e + 8;                     // latency-bound gather: prefetch a future row
    if (e2 < E_) __builtin_prefetch(hs + (size_t)src[e2] * DD, 0, 0);
  }
  float ex = expf(logits[e] - fmax_unmap(m[d]));
  if (j == 0) atomicAdd(denom + d, ex);
  atomicAdd(out + (size_t)d * DD + j, ex * hs[(size_t)s * DD + j]);
}

// ---------- normalize + bias + column sum/sumsq for BN ----------
__global__ void k_finalize_stats(float* __restrict__ out, const float* __restrict__ denom,
                                 const float* __restrict__ bias, float* __restrict__ colsum,
                                 float* __restrict__ colsq, int total) {
  __shared__ float ss[256];
  __shared__ float sq[256];
  float ls = 0.f, lq = 0.f;
  // stride is a multiple of 64 so each thread's feature index j is fixed (= tid % 64)
  for (int i = blockIdx.x * blockDim.x + threadIdx.x; i < total; i += gridDim.x * blockDim.x) {
    int n = i >> 6, j = i & 63;
    float v = out[i] / (denom[n] + SM_EPS) + bias[j];
    out[i] = v;
    ls += v; lq += v * v;
  }
  ss[threadIdx.x] = ls;
  sq[threadIdx.x] = lq;
  __syncthreads();
  if (threadIdx.x < DD) {
    float a = ss[threadIdx.x] + ss[threadIdx.x + 64] + ss[threadIdx.x + 128] + ss[threadIdx.x + 192];
    float b = sq[threadIdx.x] + sq[threadIdx.x + 64] + sq[threadIdx.x + 128] + sq[threadIdx.x + 192];
    atomicAdd(colsum + threadIdx.x, a);
    atomicAdd(colsq + threadIdx.x, b);
  }
}

// ---------- fold BN into per-feature scale/shift ----------
__global__ void k_bnparams(const float* __restrict__ colsum, const float* __restrict__ colsq,
                           const float* __restrict__ gamma, const float* __restrict__ beta,
                           float* __restrict__ sb, float invN) {
  int j = threadIdx.x;
  float mu  = colsum[j] * invN;
  float var = colsq[j] * invN - mu * mu;
  float s = gamma[j] * rsqrtf(var + BN_EPS);
  sb[j] = s;
  sb[DD + j] = beta[j] - mu * s;
}

// ---------- apply BN (+optional ELU), emit f16 activations and/or f32 output ----------
__global__ void k_apply(const float* __restrict__ pre, const float* __restrict__ sb,
                        float* __restrict__ fout, _Float16* __restrict__ h16,
                        int total, int doElu) {
  int i = blockIdx.x * blockDim.x + threadIdx.x;
  if (i >= total) return;
  int j = i & 63;
  float v = pre[i] * sb[j] + sb[DD + j];
  if (doElu) v = (v > 0.f) ? v : (expf(v) - 1.f);
  if (h16)  h16[i] = (_Float16)v;
  if (fout) fout[i] = v;
}

extern "C" void kernel_launch(void* const* d_in, const int* in_sizes, int n_in,
                              void* d_out, int out_size, void* d_ws, size_t ws_size,
                              hipStream_t stream) {
  const float* node_x = (const float*)d_in[0];
  const float* edge_x = (const float*)d_in[1];
  const int*   src    = (const int*)d_in[2];
  const int*   dst    = (const int*)d_in[3];
  // per-layer params: fc_w, fc_e_w, attn_l, attn_r, attn_e, bias
  const float* P[3][6];
  for (int l = 0; l < 3; ++l)
    for (int k = 0; k < 6; ++k)
      P[l][k] = (const float*)d_in[4 + l * 6 + k];
  const float* bn_gamma = (const float*)d_in[22];
  const float* bn_beta  = (const float*)d_in[23];

  const int N_  = in_sizes[0] / DD;
  const int E_  = in_sizes[2];
  const int TOT = N_ * DD;

  // workspace carve (256B aligned)
  char* p = (char*)d_ws;
  auto carve = [&](size_t bytes) {
    void* r = (void*)p;
    p += (bytes + 255) & ~(size_t)255;
    return r;
  };
  float*    hs     = (float*)carve((size_t)TOT * 4);
  float*    outb   = (float*)carve((size_t)TOT * 4);   // aggregation / pre-BN buffer
  _Float16* h16    = (_Float16*)carve((size_t)TOT * 2);
  float*    ee     = (float*)carve((size_t)3 * E_ * 4);
  float*    logits = (float*)carve((size_t)E_ * 4);
  float*    el     = (float*)carve((size_t)N_ * 4);
  float*    er     = (float*)carve((size_t)N_ * 4);
  unsigned* mmax   = (unsigned*)carve((size_t)N_ * 4);
  float*    denom  = (float*)carve((size_t)N_ * 4);
  _Float16* W16    = (_Float16*)carve((size_t)3 * DD * DD * 2);
  float*    ve     = (float*)carve((size_t)3 * DD * 4);
  float*    colsum = (float*)carve((size_t)DD * 4);
  float*    colsq  = (float*)carve((size_t)DD * 4);
  float*    sb     = (float*)carve((size_t)2 * DD * 4);

  const int B = 256;
  const int gElem  = (TOT + B - 1) / B;              // elementwise over N*64
  const int nTiles = (N_ + 15) / 16;
  const int gGemm  = (nTiles * 32 + B - 1) / B;
  const int gEdgeW = ((size_t)E_ * 32 + B - 1) / B;  // wave-per-edge
  const int gNodeW = ((size_t)N_ * 32 + B - 1) / B;  // wave-per-node
  const int gE     = (E_ + B - 1) / B;
  const int gAgg   = ((size_t)E_ * 64 + B - 1) / B;
  const int gStats = 1200;

  // ---- prep (once per call) ----
  k_f32_to_f16<<<gElem, B, 0, stream>>>(node_x, h16, TOT);
  k_prep_weights<<<3, B, 0, stream>>>(P[0][0], P[0][1], P[0][4],
                                      P[1][0], P[1][1], P[1][4],
                                      P[2][0], P[2][1], P[2][4], W16, ve);
  k_ee<<<gEdgeW, B, 0, stream>>>(edge_x, ve, ee, E_);

  // ---- 3 GAT layers ----
  for (int l = 0; l < 3; ++l) {
    k_gemm_wmma<<<gGemm, B, 0, stream>>>(h16, W16 + (size_t)l * DD * DD, hs, N_, nTiles);
    k_eler<<<gNodeW, B, 0, stream>>>(hs, P[l][2], P[l][3], el, er, N_);
    k_init<<<gElem, B, 0, stream>>>(outb, mmax, denom, colsum, colsq, N_, TOT);
    k_logits<<<gE, B, 0, stream>>>(el, er, ee + (size_t)l * E_, src, dst, logits, mmax, E_);
    k_aggregate<<<gAgg, B, 0, stream>>>(hs, logits, mmax, src, dst, outb, denom, E_);
    k_finalize_stats<<<gStats, B, 0, stream>>>(outb, denom, P[l][5], colsum, colsq, TOT);
    k_bnparams<<<1, DD, 0, stream>>>(colsum, colsq, bn_gamma, bn_beta, sb, 1.0f / (float)N_);
    if (l < 2) {
      k_apply<<<gElem, B, 0, stream>>>(outb, sb, nullptr, h16, TOT, 1);
    } else {
      k_apply<<<gElem, B, 0, stream>>>(outb, sb, (float*)d_out, nullptr, TOT, 0);
    }
  }
}
